// Sim_Loss_2276332666948
// MI455X (gfx1250) — matrix operand order repaired and checked by reading, not compiled, not run on hardware
//
#include <hip/hip_runtime.h>

// Problem constants (from reference)
constexpr int   Bn = 8192;
constexpr int   Vn = 1024;
constexpr int   Dn = 6;
constexpr int   Cn = 345;
constexpr int   Cp = 352;            // C padded to 22 tiles of 16
constexpr int   NTpD = Cp / 16;      // 22 n-tiles per domain
constexpr float S_LAMBDA = 1.0f;
// sig2 = 0.25 -> exp(-0.5*d2*sig2) = exp(-0.125*d2)
constexpr float EXP_SCALE = -0.125f;

typedef __attribute__((ext_vector_type(16))) __bf16 v16bf;
typedef __attribute__((ext_vector_type(8)))  __bf16 v8bf;
typedef __attribute__((ext_vector_type(8)))  float  v8f;

__device__ inline v8f wmma_bf16(v16bf a, v16bf b, v8f c) {
    // (neg_a, A, neg_b, B, c_mod, C, reuse_a, reuse_b)
    return __builtin_amdgcn_wmma_f32_16x16x32_bf16(false, a, false, b, (short)0, c,
                                                   false, false);
}

// ---- Pre-convert prototypes -> padded bf16 hi/lo, fused with p2 reduction ---------
// pHi/pLo layout: [(d*Cp + c) * Vn + v], pad columns (c >= Cn) are zero so the
// GEMM needs no clamping/branching. p2[d*Cp + c] = sum_v protos[d,c,v]^2.
__global__ __launch_bounds__(128) void pconv_kernel(const float* __restrict__ protos,
                                                    __bf16* __restrict__ pHi,
                                                    __bf16* __restrict__ pLo,
                                                    float* __restrict__ p2) {
    __shared__ float red[128];
    const int j = blockIdx.x;            // 0 .. Dn*Cp-1
    const int d = j / Cp, c = j % Cp;
    __bf16* hout = pHi + (size_t)j * Vn;
    __bf16* lout = pLo + (size_t)j * Vn;
    float acc = 0.0f;
    if (c < Cn) {
        const float* row = protos + ((size_t)d * Cn + c) * Vn;
        for (int v = threadIdx.x; v < Vn; v += 128) {
            float  x = row[v];
            __bf16 h = (__bf16)x;
            hout[v] = h;
            lout[v] = (__bf16)(x - (float)h);
            acc += x * x;
        }
    } else {
        for (int v = threadIdx.x; v < Vn; v += 128) {
            hout[v] = (__bf16)0.f;
            lout[v] = (__bf16)0.f;
        }
    }
    red[threadIdx.x] = acc;
    __syncthreads();
    for (int s = 64; s > 0; s >>= 1) {
        if (threadIdx.x < s) red[threadIdx.x] += red[threadIdx.x + s];
        __syncthreads();
    }
    if (threadIdx.x == 0) p2[j] = red[0];
}

// ---------------- f2[i] = sum_v feat[i,v]^2 ----------------------------------------
__global__ __launch_bounds__(128) void f2_kernel(const float* __restrict__ feat,
                                                 float* __restrict__ f2) {
    __shared__ float red[128];
    int i = blockIdx.x;
    const float* row = feat + (size_t)i * Vn;
    float acc = 0.0f;
    for (int v = threadIdx.x; v < Vn; v += 128) {
        float x = row[v];
        acc += x * x;
    }
    red[threadIdx.x] = acc;
    __syncthreads();
    for (int s = 64; s > 0; s >>= 1) {
        if (threadIdx.x < s) red[threadIdx.x] += red[threadIdx.x + s];
        __syncthreads();
    }
    if (threadIdx.x == 0) f2[i] = red[0];
}

// ---------------- Fused WMMA GEMM + similarity epilogue ----------------------------
// Grid: Bn/64 workgroups of 256 threads (8 waves). 64 feat rows staged once in LDS
// as bf16 hi/lo (256 KB of the 320 KB WGP LDS); B comes pre-converted from the
// L2-resident pHi/pLo arrays. Inner loop is pure loads + 3 WMMAs (hi*hi+hi*lo+lo*hi).
__global__ __launch_bounds__(256) void sim_gemm_kernel(
        const float* __restrict__ feat,
        const __bf16* __restrict__ pHi, const __bf16* __restrict__ pLo,
        const int* __restrict__ domb, const int* __restrict__ labb,
        const float* __restrict__ f2, const float* __restrict__ p2,
        float* __restrict__ simA, float* __restrict__ simB,
        float* __restrict__ simC) {
    __shared__ __bf16 Ahi[64 * Vn];          // 128 KB
    __shared__ __bf16 Alo[64 * Vn];          // 128 KB
    __shared__ float  f2s[64];
    __shared__ int    doms[64];
    __shared__ int    labs[64];

    const int tid     = threadIdx.x;
    const int rowBase = blockIdx.x * 64;

    // Stage A (f32 -> bf16 hi/lo split; one-time cost amortized over 66 n-tiles)
    for (int idx = tid; idx < 64 * Vn; idx += 256) {
        int r = idx >> 10, v = idx & (Vn - 1);
        float a = feat[(size_t)(rowBase + r) * Vn + v];
        __bf16 h = (__bf16)a;
        Ahi[idx] = h;
        Alo[idx] = (__bf16)(a - (float)h);
    }
    if (tid < 64) {
        f2s[tid]  = f2[rowBase + tid];
        doms[tid] = domb[rowBase + tid];
        labs[tid] = labb[rowBase + tid];
    }
    __syncthreads();

    const int wave  = tid >> 5;
    const int lane  = tid & 31;
    const int rsub  = wave & 3;          // which 16-row sub-tile
    const int nhalf = wave >> 2;         // which half of the 132 n-tiles
    const int rbase = rsub * 16;
    const int laneM = lane & 15;
    const int hsel  = lane >> 4;         // 0: lanes 0-15, 1: lanes 16-31
    const int aShift = hsel * 8;         // A frag: K sub-offset (ISA 16-bit A layout)
    const int bShift = hsel * 8;         // B frag: K sub-offset (same table order as A)

    const __bf16* aHiRow = &Ahi[(rbase + laneM) * Vn];
    const __bf16* aLoRow = &Alo[(rbase + laneM) * Vn];

    for (int nt = nhalf * 66; nt < nhalf * 66 + 66; ++nt) {
        const int  d      = nt / NTpD;
        const int  cb     = (nt % NTpD) * 16;
        const int  c      = cb + laneM;              // < Cp always (padded arrays)
        const bool cvalid = (c < Cn);
        const size_t bOff = (size_t)(d * Cp + c) * Vn + bShift;
        const __bf16* bHiRow = pHi + bOff;
        const __bf16* bLoRow = pLo + bOff;

        v8f acc = {0.f, 0.f, 0.f, 0.f, 0.f, 0.f, 0.f, 0.f};

#pragma unroll 2
        for (int k0 = 0; k0 < Vn; k0 += 32) {
            // A fragments from LDS (ds_load_b128 pairs)
            v8bf a0h = *(const v8bf*)(aHiRow + k0 + aShift);
            v8bf a1h = *(const v8bf*)(aHiRow + k0 + aShift + 16);
            v8bf a0l = *(const v8bf*)(aLoRow + k0 + aShift);
            v8bf a1l = *(const v8bf*)(aLoRow + k0 + aShift + 16);
            v16bf ah = __builtin_shufflevector(a0h, a1h, 0, 1, 2, 3, 4, 5, 6, 7,
                                               8, 9, 10, 11, 12, 13, 14, 15);
            v16bf al = __builtin_shufflevector(a0l, a1l, 0, 1, 2, 3, 4, 5, 6, 7,
                                               8, 9, 10, 11, 12, 13, 14, 15);
            // B fragments: pre-converted bf16 straight from L2 (two 16B chunks/lane, k+0 and k+16)
            __builtin_prefetch(bHiRow + k0 + 128, 0, 1);   // global_prefetch_b8
            v8bf b0h = *(const v8bf*)(bHiRow + k0);
            v8bf b1h = *(const v8bf*)(bHiRow + k0 + 16);
            v8bf b0l = *(const v8bf*)(bLoRow + k0);
            v8bf b1l = *(const v8bf*)(bLoRow + k0 + 16);
            v16bf bh = __builtin_shufflevector(b0h, b1h, 0, 1, 2, 3, 4, 5, 6, 7,
                                               8, 9, 10, 11, 12, 13, 14, 15);
            v16bf bl = __builtin_shufflevector(b0l, b1l, 0, 1, 2, 3, 4, 5, 6, 7,
                                               8, 9, 10, 11, 12, 13, 14, 15);
            // cross ~= hi*hi + hi*lo + lo*hi  (f32 accumulate)
            acc = wmma_bf16(ah, bh, acc);
            acc = wmma_bf16(ah, bl, acc);
            acc = wmma_bf16(al, bh, acc);
        }

        // Epilogue: C/D layout is VGPR j -> M = j + 8*hsel, N = lane&15
        const float p2v = p2[d * Cp + c];
#pragma unroll
        for (int j = 0; j < 8; ++j) {
            int   rl    = rbase + j + aShift;       // local row (0..63)
            int   i     = rowBase + rl;             // global row
            float cross = acc[j];
            float d2    = f2s[rl] + p2v - 2.0f * cross;
            d2          = fmaxf(d2, 0.0f);
            float sim   = __expf(EXP_SCALE * d2);
            int   dm    = doms[rl];
            int   lb    = labs[rl];
            bool  isA   = (d == dm) && (c == lb);
            if (isA) simA[i] = sim;                 // sim_a (pre-zeroing value)
            float sv = isA ? 0.0f : sim;            // all_sim with (dom,lab) zeroed
            if (cvalid) {
                if (c == lb) simB[i * 8 + d] = sv;  // sim_b[i,d]
                if (d == dm) simC[(size_t)i * Cp + c] = sv;  // sim_c[i,c]
            }
        }
    }
}

// ---------------- Per-row D x C max-combine -> deterministic partials --------------
__global__ __launch_bounds__(128) void combine_kernel(
        const int* __restrict__ domb, const int* __restrict__ labb,
        const float* __restrict__ simA, const float* __restrict__ simB,
        const float* __restrict__ simC, float* __restrict__ partials) {
    __shared__ float red[128];
    __shared__ float sB[8];
    __shared__ float sAsh;
    const int i   = blockIdx.x;
    const int tid = threadIdx.x;
    const int dm  = domb[i];
    const int lb  = labb[i];
    if (tid < 8) sB[tid] = (tid < Dn) ? simB[i * 8 + tid] : 0.0f;
    if (tid == 8) sAsh = simA[i];
    __syncthreads();
    const float sA = sAsh;
    float acc = 0.0f;
    for (int c = tid; c < Cn; c += 128) {
        if (c == lb) continue;
        float sc = simC[(size_t)i * Cp + c];
#pragma unroll
        for (int d = 0; d < Dn; ++d) {
            if (d == dm) continue;
            acc += fmaxf(sB[d] + sc - sA, 0.0f);
        }
    }
    red[tid] = acc;
    __syncthreads();
    for (int s = 64; s > 0; s >>= 1) {
        if (tid < s) red[tid] += red[tid + s];
        __syncthreads();
    }
    if (tid == 0) partials[i] = red[0];
}

// ---------------- Final deterministic reduction ------------------------------------
__global__ __launch_bounds__(256) void final_reduce_kernel(const float* __restrict__ partials,
                                                           float* __restrict__ out) {
    __shared__ float red[256];
    float acc = 0.0f;
    for (int i = threadIdx.x; i < Bn; i += 256) acc += partials[i];
    red[threadIdx.x] = acc;
    __syncthreads();
    for (int s = 128; s > 0; s >>= 1) {
        if (threadIdx.x < s) red[threadIdx.x] += red[threadIdx.x + s];
        __syncthreads();
    }
    if (threadIdx.x == 0) out[0] = S_LAMBDA * red[0];
}

extern "C" void kernel_launch(void* const* d_in, const int* in_sizes, int n_in,
                              void* d_out, int out_size, void* d_ws, size_t ws_size,
                              hipStream_t stream) {
    (void)in_sizes; (void)n_in; (void)out_size; (void)ws_size;
    const float* feat   = (const float*)d_in[0];   // (B,V) f32
    const float* protos = (const float*)d_in[1];   // (D,C,V) f32
    const int*   domb   = (const int*)d_in[2];     // (B,) i32
    const int*   labb   = (const int*)d_in[3];     // (B,) i32

    // Workspace layout: ~20.6 MB total
    float* ws       = (float*)d_ws;
    float* p2       = ws;                               // Dn*Cp      = 2112
    float* f2       = p2 + Dn * Cp;                     // Bn         = 8192
    float* simA     = f2 + Bn;                          // Bn         = 8192
    float* simB     = simA + Bn;                        // Bn*8       = 65536
    float* simC     = simB + Bn * 8;                    // Bn*Cp      = 2883584
    float* partials = simC + (size_t)Bn * Cp;           // Bn         = 8192
    __bf16* pHi     = (__bf16*)(partials + Bn);         // Dn*Cp*Vn bf16 = 4.33 MB
    __bf16* pLo     = pHi + (size_t)Dn * Cp * Vn;       // Dn*Cp*Vn bf16 = 4.33 MB

    pconv_kernel<<<Dn * Cp, 128, 0, stream>>>(protos, pHi, pLo, p2);
    f2_kernel<<<Bn, 128, 0, stream>>>(feat, f2);
    sim_gemm_kernel<<<Bn / 64, 256, 0, stream>>>(feat, pHi, pLo, domb, labb,
                                                 f2, p2, simA, simB, simC);
    combine_kernel<<<Bn, 128, 0, stream>>>(domb, labb, simA, simB, simC, partials);
    final_reduce_kernel<<<1, 256, 0, stream>>>(partials, (float*)d_out);
}
